// ADIAModel_60756607369341
// MI455X (gfx1250) — matrix operand order repaired and checked by reading, not compiled, not run on hardware
//
#include <hip/hip_runtime.h>
#include <math.h>

typedef _Float16 v16h __attribute__((ext_vector_type(16)));
typedef _Float16 v8h  __attribute__((ext_vector_type(8)));
typedef float    v8f  __attribute__((ext_vector_type(8)));

#define NPOS 256
#define CSTR 72   // LDS activation stride (halfs): 144B -> 16B aligned frags, conflict-free

__device__ __forceinline__ float gelu_exact(float x) {
    return 0.5f * x * (1.f + erff(x * 0.70710678118654752f));
}

// ---------------------------------------------------------------------------
// One-shot weight packer: f32 conv weights (NL,64,64,3) -> f16 fragments in
// a WMMA A-matrix lane layout with contiguous k (k = 16*(lane/16) + e), so the conv kernel can
// fetch a whole fragment with two contiguous 16B loads and zero converts.
// out index: ((L*3+t)*2+kc)*4*512 + mi*512 + lane*16 + e
// ---------------------------------------------------------------------------
__global__ __launch_bounds__(256) void pack_wmma_weights(
    const float* __restrict__ w, _Float16* __restrict__ out, int total)
{
    int idx = blockIdx.x * 256 + threadIdx.x;
    if (idx >= total) return;
    int e    = idx & 15;
    int lane = (idx >> 4) & 31;
    int mi   = (idx >> 9) & 3;
    int f    = idx >> 11;          // (L*3+t)*2+kc
    int kc   = f & 1;
    int t    = (f >> 1) % 3;
    int L    = f / 6;
    int r = e >> 1, h2 = e & 1;
    int hi = lane >> 4;
    int kk  = (hi ? 16 : 0) + (r << 1) + h2;
    int Mr  = mi * 16 + (lane & 15);
    int cin = kc * 32 + kk;
    out[idx] = (_Float16)w[((size_t)L * 64 * 64 + Mr * 64 + cin) * 3 + t];
}

// ---------------------------------------------------------------------------
// Conv stack: stem (Cin->64) + NL residual conv blocks (64->64, k=3, SAME,
// GroupNorm(8)+GELU) + mean-pool over N.  One workgroup per sample.
// Conv GEMM via v_wmma_f32_16x16x32_f16; activations in LDS (f16, [n][cin]
// with halo cols); A-frags from pre-packed f16; GN stats via LDS atomics.
// ---------------------------------------------------------------------------
template<int CIN, int NL, bool GATHER>
__global__ __launch_bounds__(256) void conv_stack_kernel(
    const float* __restrict__ edge_data, const int* __restrict__ gidx,
    const float* __restrict__ stem_w, const float* __restrict__ stem_b,
    const _Float16* __restrict__ packw, const float* __restrict__ gnw,
    const float* __restrict__ gnb, float* __restrict__ out)
{
    __shared__ __align__(16) _Float16 hbf[258 * CSTR]; // [n+1][cin]; cols 0,257 zero halo
    __shared__ float gsum[8], gsq[8];
    __shared__ float red[256];

    const int tid = threadIdx.x;
    const int m   = blockIdx.x;

    const float* chbase[CIN];
    if (!GATHER) {
        #pragma unroll
        for (int c = 0; c < CIN; ++c)
            chbase[c] = edge_data + ((size_t)m * CIN + c) * NPOS;
    } else {
        int b = m >> 3, k = m & 7;
        #pragma unroll
        for (int c = 0; c < CIN; ++c) {
            int j = c >> 3, cc = c & 7;
            int e = gidx[k * 4 + j];
            chbase[c] = edge_data + (((size_t)b * 90 + e) * 8 + cc) * (size_t)NPOS;
        }
    }

    // ---- stem ----
    for (int i = tid; i < 64 * NPOS; i += 256) {
        int n = i >> 6, ch = i & 63;
        float acc = stem_b[ch];
        #pragma unroll
        for (int c = 0; c < CIN; ++c) acc += chbase[c][n] * stem_w[c * 64 + ch];
        hbf[(n + 1) * CSTR + ch] = (_Float16)acc;
    }
    for (int i = tid; i < 128; i += 256) {
        int ch = i & 63;
        hbf[(i < 64 ? 0 : 257) * CSTR + ch] = (_Float16)0.f;
    }
    __syncthreads();

    const int wave = tid >> 5, lane = tid & 31;
    const int m0   = (wave >> 1) * 16;
    const int ngrp = wave & 1;
    const int lo16 = lane & 15, hi = lane >> 4;

    for (int L = 0; L < NL; ++L) {
        const _Float16* pwL = packw + (size_t)L * 6 * 2048;
        if (tid < 8) { gsum[tid] = 0.f; gsq[tid] = 0.f; }

        // A fragments: two contiguous 16B global loads each (pre-packed layout)
        v16h afrag[6];
        #pragma unroll
        for (int t = 0; t < 3; ++t) {
            #pragma unroll
            for (int kc = 0; kc < 2; ++kc) {
                const _Float16* ap_ = pwL + (((t * 2 + kc) * 4 + (m0 >> 4)) * 512) + lane * 16;
                union { v16h v; v8h h[2]; } u;
                u.h[0] = *(const v8h*)(ap_);
                u.h[1] = *(const v8h*)(ap_ + 8);
                afrag[t * 2 + kc] = u.v;
            }
        }

        // GEMM: 8 tiles of 16x16, each = 6 chained WMMAs over (tap, K-chunk)
        v8f acc[8];
        #pragma unroll
        for (int i = 0; i < 8; ++i) {
            int n0 = (ngrp * 8 + i) * 16;
            v8f c = {0.f, 0.f, 0.f, 0.f, 0.f, 0.f, 0.f, 0.f};
            #pragma unroll
            for (int t = 0; t < 3; ++t) {
                #pragma unroll
                for (int kc = 0; kc < 2; ++kc) {
                    // B 32x16: lane holds col N=lane&15; lanes0-15 K=0..15,
                    // lanes16-31 K=16..31; 16 contiguous halfs, 16B aligned.
                    const _Float16* p = &hbf[(n0 + lo16 + t) * CSTR + kc * 32 + hi * 16];
                    union { v16h v; v8h h[2]; } u;
                    u.h[0] = *(const v8h*)(p);
                    u.h[1] = *(const v8h*)(p + 8);
                    c = __builtin_amdgcn_wmma_f32_16x16x32_f16(
                            false, afrag[t * 2 + kc], false, u.v, (short)0, c, false, false);
                }
            }
            acc[i] = c;
        }
        __syncthreads();   // gsum zeroed; all B-reads of hbf complete

        // GroupNorm stats
        {
            int g = (m0 >> 3) + hi;
            float s = 0.f, q = 0.f;
            #pragma unroll
            for (int i = 0; i < 8; ++i)
                #pragma unroll
                for (int r = 0; r < 8; ++r) { float v = acc[i][r]; s += v; q += v * v; }
            atomicAdd(&gsum[g], s);
            atomicAdd(&gsq[g], q);
        }
        __syncthreads();

        // GN + GELU + residual
        {
            int g = (m0 >> 3) + hi;
            float mean = gsum[g] * (1.f / 2048.f);
            float var  = gsq[g] * (1.f / 2048.f) - mean * mean;
            float rinv = rsqrtf(var + 1e-5f);
            #pragma unroll
            for (int i = 0; i < 8; ++i) {
                int n0  = (ngrp * 8 + i) * 16;
                int col = n0 + lo16 + 1;
                #pragma unroll
                for (int r = 0; r < 8; ++r) {
                    int Mr   = m0 + hi * 8 + r;
                    float v  = (acc[i][r] - mean) * rinv * gnw[L * 64 + Mr] + gnb[L * 64 + Mr];
                    float a2 = gelu_exact(v);
                    float ho = (float)hbf[col * CSTR + Mr];
                    hbf[col * CSTR + Mr] = (_Float16)(ho + a2);
                }
            }
        }
        __syncthreads();
    }

    // ---- mean over N ----
    {
        int ch = tid & 63, part = tid >> 6;
        float s = 0.f;
        for (int n = part * 64; n < part * 64 + 64; ++n) s += (float)hbf[(n + 1) * CSTR + ch];
        red[part * 64 + ch] = s;
        __syncthreads();
        if (tid < 64) {
            float tot = red[tid] + red[64 + tid] + red[128 + tid] + red[192 + tid];
            out[(size_t)m * 64 + tid] = tot * (1.f / 256.f);
        }
    }
}

// ---------------------------------------------------------------------------
__global__ __launch_bounds__(64) void edge_merge_kernel(
    const float* __restrict__ raw, const int* __restrict__ etypes,
    const float* __restrict__ type_emb, const float* __restrict__ W,
    const float* __restrict__ bias, const float* __restrict__ lnw,
    const float* __restrict__ lnb, float* __restrict__ out)
{
    __shared__ float xr[128];
    __shared__ float yr[64];
    int m = blockIdx.x, t = threadIdx.x;
    int ty = etypes[m];
    xr[t]      = raw[(size_t)m * 64 + t];
    xr[64 + t] = type_emb[ty * 64 + t];
    __syncthreads();
    float acc = bias[t];
    for (int c = 0; c < 128; ++c) acc += xr[c] * W[c * 64 + t];
    yr[t] = acc;
    __syncthreads();
    float s = 0.f; for (int i = 0; i < 64; ++i) s += yr[i];
    float mean = s * (1.f / 64.f);
    float q = 0.f; for (int i = 0; i < 64; ++i) { float d = yr[i] - mean; q += d * d; }
    float rv = rsqrtf(q * (1.f / 64.f) + 1e-5f);
    float v = (acc - mean) * rv * lnw[t] + lnb[t];
    out[(size_t)m * 64 + t] = gelu_exact(v);
}

// ---------------------------------------------------------------------------
__global__ __launch_bounds__(256) void attn_kernel(
    float* __restrict__ emb, const int* __restrict__ srel,
    const unsigned char* __restrict__ emask,
    const float* qw, const float* qb, const float* kw, const float* kb,
    const float* vw, const float* vb, const float* ow, const float* ob,
    const float* structp, const float* ln1w, const float* ln1b,
    const float* ff1w, const float* ff1b, const float* ff2w, const float* ff2b,
    const float* ln2w, const float* ln2b,
    float* __restrict__ hidws, float* __restrict__ ybufws)
{
    __shared__ float xs[90 * 64];
    __shared__ float Qh[90 * 16], Kh[90 * 16], Vh[90 * 16];
    __shared__ _Float16 sc[90 * 90];
    const int b = blockIdx.x, t = threadIdx.x;
    float* ybuf = ybufws + (size_t)b * 90 * 64;
    float* hid  = hidws  + (size_t)b * 90 * 256;

    for (int i = t; i < 90 * 64; i += 256) xs[i] = emb[(size_t)b * 90 * 64 + i];
    __syncthreads();

    for (int h = 0; h < 4; ++h) {
        for (int i = t; i < 90 * 16; i += 256) {
            int e = i >> 4, d = i & 15, col = h * 16 + d;
            float aq = qb[col], ak = kb[col], av = vb[col];
            for (int c2 = 0; c2 < 64; ++c2) {
                float xv = xs[e * 64 + c2];
                aq += xv * qw[c2 * 64 + col];
                ak += xv * kw[c2 * 64 + col];
                av += xv * vw[c2 * 64 + col];
            }
            Qh[i] = aq; Kh[i] = ak; Vh[i] = av;
        }
        __syncthreads();
        for (int i = t; i < 90 * 90; i += 256) {
            int qi = i / 90, ki = i - qi * 90;
            float s = 0.f;
            #pragma unroll
            for (int d2 = 0; d2 < 16; ++d2) s += Qh[qi * 16 + d2] * Kh[ki * 16 + d2];
            s *= 0.25f;
            int rel = srel[((size_t)b * 90 + qi) * 90 + ki];
            s += structp[rel * 4 + h];
            if (emask[b * 90 + ki] == 0) s = -3.0e4f;
            sc[i] = (_Float16)s;
        }
        __syncthreads();
        if (t < 90) {
            float mx = -1e30f;
            for (int k2 = 0; k2 < 90; ++k2) mx = fmaxf(mx, (float)sc[t * 90 + k2]);
            float ssum = 0.f;
            for (int k2 = 0; k2 < 90; ++k2) {
                float ev = __expf((float)sc[t * 90 + k2] - mx);
                ssum += ev; sc[t * 90 + k2] = (_Float16)ev;
            }
            float inv = (ssum > 0.f) ? 1.f / ssum : 0.f;
            for (int k2 = 0; k2 < 90; ++k2)
                sc[t * 90 + k2] = (_Float16)((float)sc[t * 90 + k2] * inv);
        }
        __syncthreads();
        for (int i = t; i < 90 * 16; i += 256) {
            int e = i >> 4, d = i & 15;
            float s = 0.f;
            for (int k2 = 0; k2 < 90; ++k2) s += (float)sc[e * 90 + k2] * Vh[k2 * 16 + d];
            Qh[i] = s;
        }
        __syncthreads();
        for (int i = t; i < 90 * 64; i += 256) {
            int e = i >> 6, d = i & 63;
            float s = 0.f;
            #pragma unroll
            for (int dd = 0; dd < 16; ++dd) s += Qh[e * 16 + dd] * ow[(h * 16 + dd) * 64 + d];
            if (h == 0) ybuf[i] = xs[i] + ob[d] + s; else ybuf[i] += s;
        }
        __syncthreads();
    }
    if (t < 90) {
        float s = 0.f; for (int d = 0; d < 64; ++d) s += ybuf[t * 64 + d];
        float mean = s * (1.f / 64.f);
        float q = 0.f; for (int d = 0; d < 64; ++d) { float dd = ybuf[t * 64 + d] - mean; q += dd * dd; }
        float rv = rsqrtf(q * (1.f / 64.f) + 1e-5f);
        for (int d = 0; d < 64; ++d)
            xs[t * 64 + d] = (ybuf[t * 64 + d] - mean) * rv * ln1w[d] + ln1b[d];
    }
    __syncthreads();
    for (int i = t; i < 90 * 256; i += 256) {
        int e = i >> 8, j = i & 255;
        float s = ff1b[j];
        for (int c2 = 0; c2 < 64; ++c2) s += xs[e * 64 + c2] * ff1w[c2 * 256 + j];
        hid[i] = gelu_exact(s);
    }
    __syncthreads();
    for (int i = t; i < 90 * 64; i += 256) {
        int e = i >> 6, d = i & 63;
        float s = ff2b[d];
        for (int c2 = 0; c2 < 256; ++c2) s += hid[e * 256 + c2] * ff2w[c2 * 64 + d];
        ybuf[i] = xs[i] + s;
    }
    __syncthreads();
    if (t < 90) {
        float s = 0.f; for (int d = 0; d < 64; ++d) s += ybuf[t * 64 + d];
        float mean = s * (1.f / 64.f);
        float q = 0.f; for (int d = 0; d < 64; ++d) { float dd = ybuf[t * 64 + d] - mean; q += dd * dd; }
        float rv = rsqrtf(q * (1.f / 64.f) + 1e-5f);
        for (int d = 0; d < 64; ++d)
            emb[(size_t)b * 90 * 64 + t * 64 + d] =
                (ybuf[t * 64 + d] - mean) * rv * ln2w[d] + ln2b[d];
    }
}

__global__ __launch_bounds__(256) void edge_head_kernel(
    const float* __restrict__ emb, const float* __restrict__ w,
    const float* __restrict__ bb, float* __restrict__ out)
{
    int i = blockIdx.x * 256 + threadIdx.x;
    if (i >= 5760 * 2) return;
    int m = i >> 1, o = i & 1;
    float s = bb[o];
    for (int c = 0; c < 64; ++c) s += emb[(size_t)m * 64 + c] * w[c * 2 + o];
    out[i] = s;
}

__global__ __launch_bounds__(64) void node_merge_kernel(
    const float* __restrict__ emb, const int* __restrict__ gidx,
    const float* __restrict__ W, const float* __restrict__ bb,
    const float* __restrict__ lnw, const float* __restrict__ lnb,
    float* __restrict__ out)
{
    __shared__ float xr[256];
    __shared__ float yr[64];
    int m = blockIdx.x, t = threadIdx.x;
    int b = m >> 3, k = m & 7;
    for (int j = 0; j < 4; ++j) {
        int e = gidx[k * 4 + j];
        xr[j * 64 + t] = emb[((size_t)b * 90 + e) * 64 + t];
    }
    __syncthreads();
    float acc = bb[t];
    for (int c = 0; c < 256; ++c) acc += xr[c] * W[c * 64 + t];
    yr[t] = acc;
    __syncthreads();
    float s = 0.f; for (int i = 0; i < 64; ++i) s += yr[i];
    float mean = s * (1.f / 64.f);
    float q = 0.f; for (int i = 0; i < 64; ++i) { float d = yr[i] - mean; q += d * d; }
    float rv = rsqrtf(q * (1.f / 64.f) + 1e-5f);
    float v = (acc - mean) * rv * lnw[t] + lnb[t];
    out[(size_t)m * 64 + t] = gelu_exact(v);
}

__global__ __launch_bounds__(64) void node_head_kernel(
    const float* __restrict__ v11, const float* __restrict__ nc,
    const float* __restrict__ gw, const float* __restrict__ gb,
    const float* __restrict__ cw, const float* __restrict__ cb,
    float* __restrict__ out)
{
    __shared__ float f[64];
    int m = blockIdx.x, t = threadIdx.x;
    float s = gb[t];
    for (int c = 0; c < 64; ++c) s += nc[(size_t)m * 64 + c] * gw[c * 64 + t];
    f[t] = v11[(size_t)m * 64 + t] + s;
    __syncthreads();
    if (t < 8) {
        float s2 = cb[t];
        for (int c = 0; c < 64; ++c) s2 += f[c] * cw[c * 8 + t];
        out[11520 + m * 8 + t] = s2;
    }
}

// ---------------------------------------------------------------------------
extern "C" void kernel_launch(void* const* d_in, const int* in_sizes, int n_in,
                              void* d_out, int out_size, void* d_ws, size_t ws_size,
                              hipStream_t stream)
{
    (void)in_sizes; (void)n_in; (void)out_size; (void)ws_size;
    const float* edge_data = (const float*)d_in[0];
    const int*   edge_types = (const int*)d_in[1];
    const int*   struct_rel = (const int*)d_in[2];
    const unsigned char* edge_mask = (const unsigned char*)d_in[3];
    const int*   gather_idx = (const int*)d_in[4];
    int pi = 5;
    const float* ex_stem_w = (const float*)d_in[pi++];
    const float* ex_stem_b = (const float*)d_in[pi++];
    const float* ex_conv_w = (const float*)d_in[pi++];
    const float* ex_gn_w   = (const float*)d_in[pi++];
    const float* ex_gn_b   = (const float*)d_in[pi++];
    const float* type_emb  = (const float*)d_in[pi++];
    const float* em_w      = (const float*)d_in[pi++];
    const float* em_b      = (const float*)d_in[pi++];
    const float* em_ln_w   = (const float*)d_in[pi++];
    const float* em_ln_b   = (const float*)d_in[pi++];
    const float* ap[2][17];
    for (int l = 0; l < 2; ++l)
        for (int j = 0; j < 17; ++j) ap[l][j] = (const float*)d_in[pi++];
    const float* eh_w      = (const float*)d_in[pi++];
    const float* eh_b      = (const float*)d_in[pi++];
    const float* nc_stem_w = (const float*)d_in[pi++];
    const float* nc_stem_b = (const float*)d_in[pi++];
    const float* nc_conv_w = (const float*)d_in[pi++];
    const float* nc_gn_w   = (const float*)d_in[pi++];
    const float* nc_gn_b   = (const float*)d_in[pi++];
    const float* nm_w      = (const float*)d_in[pi++];
    const float* nm_b      = (const float*)d_in[pi++];
    const float* nm_ln_w   = (const float*)d_in[pi++];
    const float* nm_ln_b   = (const float*)d_in[pi++];
    const float* gate_w    = (const float*)d_in[pi++];
    const float* gate_b    = (const float*)d_in[pi++];
    const float* cls_w     = (const float*)d_in[pi++];
    const float* cls_b     = (const float*)d_in[pi++];

    float* out = (float*)d_out;

    // workspace layout (floats; all region starts stay 32B aligned)
    float*      emb_raw  = (float*)d_ws;                 // 5760*64
    float*      emb      = emb_raw + 5760 * 64;          // 5760*64
    float*      hid      = emb + 5760 * 64;              // 64*90*256
    float*      ybuf     = hid + 64 * 90 * 256;          // 64*90*64
    float*      ncout    = ybuf + 64 * 90 * 64;          // 512*64
    float*      v11      = ncout + 512 * 64;             // 512*64
    _Float16*   packed_ex = (_Float16*)(v11 + 512 * 64); // 5*6*2048 halfs
    _Float16*   packed_nc = packed_ex + 5 * 6 * 2048;    // 3*6*2048 halfs

    // 0. pack conv weights into WMMA A-fragment layout (f16)
    {
        int tot_ex = 5 * 6 * 2048;
        int tot_nc = 3 * 6 * 2048;
        pack_wmma_weights<<<(tot_ex + 255) / 256, 256, 0, stream>>>(ex_conv_w, packed_ex, tot_ex);
        pack_wmma_weights<<<(tot_nc + 255) / 256, 256, 0, stream>>>(nc_conv_w, packed_nc, tot_nc);
    }

    // 1. edge feature extractor (WMMA conv stack)
    conv_stack_kernel<8, 5, false><<<5760, 256, 0, stream>>>(
        edge_data, nullptr, ex_stem_w, ex_stem_b, packed_ex, ex_gn_w, ex_gn_b, emb_raw);

    // 2. merge with type embedding
    edge_merge_kernel<<<5760, 64, 0, stream>>>(
        emb_raw, edge_types, type_emb, em_w, em_b, em_ln_w, em_ln_b, emb);

    // 3. 2x structural attention (in place)
    for (int l = 0; l < 2; ++l) {
        const float* const* p = ap[l];
        attn_kernel<<<64, 256, 0, stream>>>(
            emb, struct_rel, edge_mask,
            p[0], p[1], p[2], p[3], p[4], p[5], p[6], p[7], p[8],
            p[9], p[10], p[11], p[12], p[13], p[14], p[15], p[16],
            hid, ybuf);
    }

    // 4. edge head -> out[0 .. 11520)
    edge_head_kernel<<<(5760 * 2 + 255) / 256, 256, 0, stream>>>(emb, eh_w, eh_b, out);

    // 5. node path
    node_merge_kernel<<<512, 64, 0, stream>>>(emb, gather_idx, nm_w, nm_b, nm_ln_w, nm_ln_b, v11);
    conv_stack_kernel<32, 3, true><<<512, 256, 0, stream>>>(
        edge_data, gather_idx, nc_stem_w, nc_stem_b, packed_nc, nc_gn_w, nc_gn_b, ncout);
    node_head_kernel<<<512, 64, 0, stream>>>(v11, ncout, gate_w, gate_b, cls_w, cls_b, out);
}
